// S5SSM_81939386073114
// MI455X (gfx1250) — compile-verified, matrix operand order checked by reading, not run on hardware
//
#include <hip/hip_runtime.h>
#include <hip/hip_bf16.h>

// Problem constants (from the reference)
#define LSEQ 16384
#define HDIM 1024
#define PDIM 512
#define BLK  128              // scan block length
#define NBLK (LSEQ / BLK)     // 128 scan blocks

typedef __attribute__((ext_vector_type(16))) __bf16 v16bf;
typedef __attribute__((ext_vector_type(8)))  float  v8f;

union V16U {
    v16bf v;
    unsigned short s[16];
    uint4 q[2];
};

// ---------- helpers ----------

__device__ __forceinline__ unsigned short f2bf(float f) {
    unsigned int u = __float_as_uint(f);
    unsigned int r = 0x7FFFu + ((u >> 16) & 1u);   // round to nearest even
    u += r;
    return (unsigned short)(u >> 16);
}

// Load a 16-element bf16 operand slice for wave32 WMMA 16x16x32 from global.
// Source is row-major with row stride `ld` (elements). K layout per CDNA5 ISA:
//   lane<16 : elems 0..7 = K k0+0..7,  elems 8..15 = K k0+16..23
//   lane>=16: elems 0..7 = K k0+8..15, elems 8..15 = K k0+24..31
__device__ __forceinline__ v16bf load_bf16_tile(const unsigned short* __restrict__ base,
                                                int ld, int row, int k0, int hi) {
    V16U t;
    const unsigned short* p0 = base + (size_t)row * ld + k0 + (hi ? 8 : 0);
    const unsigned short* p1 = base + (size_t)row * ld + k0 + (hi ? 24 : 16);
    t.q[0] = *(const uint4*)p0;
    t.q[1] = *(const uint4*)p1;
    return t.v;
}

// Same operand slice from an LDS-staged 16x32 tile (row stride 32 elements).
__device__ __forceinline__ v16bf lds_bf16_tile(const unsigned short* sp, int lo, int hi) {
    V16U t;
    t.q[0] = *(const uint4*)(sp + lo * 32 + (hi ? 8 : 0));
    t.q[1] = *(const uint4*)(sp + lo * 32 + (hi ? 24 : 16));
    return t.v;
}

// ---------- precompute kernels ----------

// One block, PDIM threads: Lambda_bar = exp(Lambda*dt); factor = (Lambda_bar-1)/Lambda
__global__ void s5_disc_kernel(const float* __restrict__ Lre, const float* __restrict__ Lim,
                               const float* __restrict__ log_step,
                               float* __restrict__ lam_re, float* __restrict__ lam_im,
                               float* __restrict__ fac_re, float* __restrict__ fac_im) {
    int p = threadIdx.x;
    if (p >= PDIM) return;
    float dt  = __expf(log_step[p]);
    float are = Lre[p], aim = Lim[p];
    float mre = are * dt, mim = aim * dt;
    float mag = __expf(mre);
    float lbr = mag * __cosf(mim);
    float lbi = mag * __sinf(mim);
    lam_re[p] = lbr;
    lam_im[p] = lbi;
    float nre = lbr - 1.0f, nim = lbi;
    float den = are * are + aim * aim;
    fac_re[p] = (nre * are + nim * aim) / den;
    fac_im[p] = (nim * are - nre * aim) / den;
}

// u (L,H) f32 -> bf16, 8 elements/thread, b128 in / b128 out
__global__ void s5_u2bf_kernel(const float* __restrict__ u, unsigned short* __restrict__ ubf) {
    size_t i = ((size_t)blockIdx.x * blockDim.x + threadIdx.x) * 8;
    float4 f0 = *(const float4*)(u + i);
    float4 f1 = *(const float4*)(u + i + 4);
    uint4 o;
    o.x = ((unsigned)f2bf(f0.y) << 16) | f2bf(f0.x);
    o.y = ((unsigned)f2bf(f0.w) << 16) | f2bf(f0.z);
    o.z = ((unsigned)f2bf(f1.y) << 16) | f2bf(f1.x);
    o.w = ((unsigned)f2bf(f1.w) << 16) | f2bf(f1.z);
    *(uint4*)(ubf + i) = o;
}

// B_bar = factor * (B_re + i B_im), emitted as bf16 (P,H) row-major
__global__ void s5_bbar_kernel(const float* __restrict__ Bre, const float* __restrict__ Bim,
                               const float* __restrict__ fac_re, const float* __restrict__ fac_im,
                               unsigned short* __restrict__ Bre_bf, unsigned short* __restrict__ Bim_bf) {
    int idx = blockIdx.x * blockDim.x + threadIdx.x;
    if (idx >= PDIM * HDIM) return;
    int p = idx / HDIM;
    float fr = fac_re[p], fi = fac_im[p];
    float br = Bre[idx], bi = Bim[idx];
    Bre_bf[idx] = f2bf(fr * br - fi * bi);
    Bim_bf[idx] = f2bf(fr * bi + fi * br);
}

// C_re -> bf16, C_im -> -bf16 (so GEMM2 can use a single accumulator)
__global__ void s5_cbf_kernel(const float* __restrict__ Cre, const float* __restrict__ Cim,
                              unsigned short* __restrict__ Cre_bf, unsigned short* __restrict__ Cng_bf) {
    int idx = blockIdx.x * blockDim.x + threadIdx.x;
    if (idx >= HDIM * PDIM) return;
    Cre_bf[idx] = f2bf(Cre[idx]);
    Cng_bf[idx] = f2bf(-Cim[idx]);
}

// ---------- GEMM 1: Bu = u @ B_bar^T  (L,H)x(H,P) -> (L,P), re & im ----------
// Block: 8 waves, covers 512 rows x 16 cols. B tiles (16x32 re+im) are staged
// into double-buffered LDS once per block per K-step; each wave reads them via
// ds_load_b128. A operand streams from global. Per wave per K-step: 8 WMMAs.

__global__ __launch_bounds__(256) void s5_gemm1_kernel(
        const unsigned short* __restrict__ u_bf,      // (L,H) bf16
        const unsigned short* __restrict__ Bre_bf,    // (P,H) bf16
        const unsigned short* __restrict__ Bim_bf,    // (P,H) bf16
        float* __restrict__ BuRe, float* __restrict__ BuIm) {  // (L,P) f32
    __shared__ __align__(16) unsigned short sRe[2][512];
    __shared__ __align__(16) unsigned short sIm[2][512];

    const int t    = threadIdx.x;
    const int lane = t & 31;
    const int wave = t >> 5;
    const int lo = lane & 15, hi = lane >> 4;
    const int m0 = blockIdx.x * 512 + wave * 64;        // wave covers rows m0..m0+63
    const int n0 = blockIdx.y * 16;                     // over P

    // cooperative stage: 256 threads x 1 dword = 512 bf16 per matrix
    const int crow = t >> 4;            // 0..15
    const int ckk  = (t & 15) * 2;      // 0,2,..,30
    const size_t csrc = (size_t)(n0 + crow) * HDIM + ckk;

    v8f cre[4] = {};
    v8f cim[4] = {};

    // prologue: stage K-step 0 into buffer 0
    *(unsigned int*)(&sRe[0][crow * 32 + ckk]) = *(const unsigned int*)(Bre_bf + csrc);
    *(unsigned int*)(&sIm[0][crow * 32 + ckk]) = *(const unsigned int*)(Bim_bf + csrc);
    __syncthreads();

    const int KT = HDIM / 32;
    for (int kt = 0; kt < KT; ++kt) {
        const int cur = kt & 1;
        const int k0 = kt * 32;
        if (kt + 1 < KT) {
            const int nxt = cur ^ 1;
            *(unsigned int*)(&sRe[nxt][crow * 32 + ckk]) = *(const unsigned int*)(Bre_bf + csrc + k0 + 32);
            *(unsigned int*)(&sIm[nxt][crow * 32 + ckk]) = *(const unsigned int*)(Bim_bf + csrc + k0 + 32);
            __builtin_prefetch(u_bf + (size_t)(m0 + lo) * HDIM + k0 + 32, 0, 0);
        }
        v16bf br = lds_bf16_tile(sRe[cur], lo, hi);
        v16bf bi = lds_bf16_tile(sIm[cur], lo, hi);
#pragma unroll
        for (int mt = 0; mt < 4; ++mt) {
            v16bf a = load_bf16_tile(u_bf, HDIM, m0 + mt * 16 + lo, k0, hi);
            cre[mt] = __builtin_amdgcn_wmma_f32_16x16x32_bf16(false, a, false, br, (short)0, cre[mt], false, false);
            cim[mt] = __builtin_amdgcn_wmma_f32_16x16x32_bf16(false, a, false, bi, (short)0, cim[mt], false, false);
        }
        __syncthreads();
    }

#pragma unroll
    for (int mt = 0; mt < 4; ++mt) {
#pragma unroll
        for (int r = 0; r < 8; ++r) {
            int row = m0 + mt * 16 + r + hi * 8;
            BuRe[(size_t)row * PDIM + n0 + lo] = cre[mt][r];
            BuIm[(size_t)row * PDIM + n0 + lo] = cim[mt][r];
        }
    }
}

// ---------- scan phase 1: local block scans (in place), emit carries ----------

__global__ void s5_scan1_kernel(const float* __restrict__ lam_re, const float* __restrict__ lam_im,
                                float* __restrict__ BuRe, float* __restrict__ BuIm,
                                float* __restrict__ car_re, float* __restrict__ car_im) {
    int t = blockIdx.x * blockDim.x + threadIdx.x;   // PDIM * NBLK threads
    int p = t & (PDIM - 1);
    int b = t >> 9;                                  // PDIM = 512 = 2^9
    float lr = lam_re[p], li = lam_im[p];
    float xr = 0.f, xi = 0.f;
    int base = b * BLK;
    for (int i = 0; i < BLK; ++i) {
        size_t idx = (size_t)(base + i) * PDIM + p;
        float br = BuRe[idx], bi = BuIm[idx];
        float nr = lr * xr - li * xi + br;
        float ni = lr * xi + li * xr + bi;
        xr = nr; xi = ni;
        BuRe[idx] = xr;
        BuIm[idx] = xi;
    }
    car_re[(size_t)b * PDIM + p] = xr;
    car_im[(size_t)b * PDIM + p] = xi;
}

// ---------- scan phase 2: per-channel scan of block carries -> block start states ----------

__global__ void s5_scan2_kernel(const float* __restrict__ lam_re, const float* __restrict__ lam_im,
                                const float* __restrict__ car_re, const float* __restrict__ car_im,
                                float* __restrict__ st_re, float* __restrict__ st_im) {
    int p = blockIdx.x * blockDim.x + threadIdx.x;
    if (p >= PDIM) return;
    float lr = lam_re[p], li = lam_im[p];
    // A = lambda_bar^BLK
    float ar = 1.f, ai = 0.f;
    for (int i = 0; i < BLK; ++i) {
        float nr = ar * lr - ai * li;
        float ni = ar * li + ai * lr;
        ar = nr; ai = ni;
    }
    float sr = 0.f, si = 0.f;
    st_re[p] = 0.f;
    st_im[p] = 0.f;
    for (int b = 1; b < NBLK; ++b) {
        float cr = car_re[(size_t)(b - 1) * PDIM + p];
        float ci = car_im[(size_t)(b - 1) * PDIM + p];
        float nr = ar * sr - ai * si + cr;
        float ni = ar * si + ai * sr + ci;
        sr = nr; si = ni;
        st_re[(size_t)b * PDIM + p] = sr;
        st_im[(size_t)b * PDIM + p] = si;
    }
}

// ---------- scan phase 3: fixup x = lam^(t+1)*s_b + local, emit bf16 ----------

__global__ void s5_scan3_kernel(const float* __restrict__ lam_re, const float* __restrict__ lam_im,
                                const float* __restrict__ BuRe, const float* __restrict__ BuIm,
                                const float* __restrict__ st_re, const float* __restrict__ st_im,
                                unsigned short* __restrict__ xre_bf, unsigned short* __restrict__ xim_bf) {
    int t = blockIdx.x * blockDim.x + threadIdx.x;
    int p = t & (PDIM - 1);
    int b = t >> 9;
    float lr = lam_re[p], li = lam_im[p];
    float sr = st_re[(size_t)b * PDIM + p];
    float si = st_im[(size_t)b * PDIM + p];
    float pr = lr, pi = li;                          // lambda^(1)
    int base = b * BLK;
    for (int i = 0; i < BLK; ++i) {
        size_t idx = (size_t)(base + i) * PDIM + p;
        float xr = pr * sr - pi * si + BuRe[idx];
        float xi = pr * si + pi * sr + BuIm[idx];
        xre_bf[idx] = f2bf(xr);
        xim_bf[idx] = f2bf(xi);
        float nr = pr * lr - pi * li;
        float ni = pr * li + pi * lr;
        pr = nr; pi = ni;
    }
}

// ---------- GEMM 2: y = 2*(x_re@C_re^T + x_im@(-C_im)^T) + D*u ----------
// Same structure as GEMM1: 8 waves/block, LDS-staged double-buffered C tiles.

__global__ __launch_bounds__(256) void s5_gemm2_kernel(
        const unsigned short* __restrict__ xre_bf,    // (L,P) bf16
        const unsigned short* __restrict__ xim_bf,    // (L,P) bf16
        const unsigned short* __restrict__ Cre_bf,    // (H,P) bf16
        const unsigned short* __restrict__ Cng_bf,    // (H,P) bf16 = -C_im
        const float* __restrict__ u,                  // (L,H) f32
        const float* __restrict__ Dvec,               // (H,)
        float* __restrict__ out) {                    // (L,H) f32
    __shared__ __align__(16) unsigned short sRe[2][512];
    __shared__ __align__(16) unsigned short sIm[2][512];

    const int t    = threadIdx.x;
    const int lane = t & 31;
    const int wave = t >> 5;
    const int lo = lane & 15, hi = lane >> 4;
    const int m0 = blockIdx.x * 512 + wave * 64;        // over L
    const int n0 = blockIdx.y * 16;                     // over H

    const int crow = t >> 4;
    const int ckk  = (t & 15) * 2;
    const size_t csrc = (size_t)(n0 + crow) * PDIM + ckk;

    v8f acc[4] = {};

    *(unsigned int*)(&sRe[0][crow * 32 + ckk]) = *(const unsigned int*)(Cre_bf + csrc);
    *(unsigned int*)(&sIm[0][crow * 32 + ckk]) = *(const unsigned int*)(Cng_bf + csrc);
    __syncthreads();

    const int KT = PDIM / 32;
    for (int kt = 0; kt < KT; ++kt) {
        const int cur = kt & 1;
        const int k0 = kt * 32;
        if (kt + 1 < KT) {
            const int nxt = cur ^ 1;
            *(unsigned int*)(&sRe[nxt][crow * 32 + ckk]) = *(const unsigned int*)(Cre_bf + csrc + k0 + 32);
            *(unsigned int*)(&sIm[nxt][crow * 32 + ckk]) = *(const unsigned int*)(Cng_bf + csrc + k0 + 32);
            __builtin_prefetch(xre_bf + (size_t)(m0 + lo) * PDIM + k0 + 32, 0, 0);
            __builtin_prefetch(xim_bf + (size_t)(m0 + lo) * PDIM + k0 + 32, 0, 0);
        }
        v16bf br = lds_bf16_tile(sRe[cur], lo, hi);
        v16bf bi = lds_bf16_tile(sIm[cur], lo, hi);
#pragma unroll
        for (int mt = 0; mt < 4; ++mt) {
            int arow = m0 + mt * 16 + lo;
            v16bf ar = load_bf16_tile(xre_bf, PDIM, arow, k0, hi);
            v16bf ai = load_bf16_tile(xim_bf, PDIM, arow, k0, hi);
            acc[mt] = __builtin_amdgcn_wmma_f32_16x16x32_bf16(false, ar, false, br, (short)0, acc[mt], false, false);
            acc[mt] = __builtin_amdgcn_wmma_f32_16x16x32_bf16(false, ai, false, bi, (short)0, acc[mt], false, false);
        }
        __syncthreads();
    }

    int col = n0 + lo;
    float d = Dvec[col];
#pragma unroll
    for (int mt = 0; mt < 4; ++mt) {
#pragma unroll
        for (int r = 0; r < 8; ++r) {
            int row = m0 + mt * 16 + r + hi * 8;
            size_t idx = (size_t)row * HDIM + col;
            out[idx] = 2.0f * acc[mt][r] + d * u[idx];
        }
    }
}

// ---------- host launcher ----------

extern "C" void kernel_launch(void* const* d_in, const int* in_sizes, int n_in,
                              void* d_out, int out_size, void* d_ws, size_t ws_size,
                              hipStream_t stream) {
    const float* u        = (const float*)d_in[0];   // (L,H)
    const float* Lre      = (const float*)d_in[1];   // (P,)
    const float* Lim      = (const float*)d_in[2];   // (P,)
    const float* Bre      = (const float*)d_in[3];   // (P,H)
    const float* Bim      = (const float*)d_in[4];   // (P,H)
    const float* Cre      = (const float*)d_in[5];   // (H,P)
    const float* Cim      = (const float*)d_in[6];   // (H,P)
    const float* Dvec     = (const float*)d_in[7];   // (H,)
    const float* log_step = (const float*)d_in[8];   // (P,1)
    float* out = (float*)d_out;

    // workspace carve-out (256B aligned slices)
    char* w = (char*)d_ws;
    size_t off = 0;
    auto carve = [&](size_t bytes) -> void* {
        void* r = w + off;
        off = (off + bytes + 255) & ~(size_t)255;
        return r;
    };
    float* lam_re = (float*)carve(PDIM * sizeof(float));
    float* lam_im = (float*)carve(PDIM * sizeof(float));
    float* fac_re = (float*)carve(PDIM * sizeof(float));
    float* fac_im = (float*)carve(PDIM * sizeof(float));
    unsigned short* Bre_bf = (unsigned short*)carve((size_t)PDIM * HDIM * 2);
    unsigned short* Bim_bf = (unsigned short*)carve((size_t)PDIM * HDIM * 2);
    unsigned short* Cre_bf = (unsigned short*)carve((size_t)HDIM * PDIM * 2);
    unsigned short* Cng_bf = (unsigned short*)carve((size_t)HDIM * PDIM * 2);
    unsigned short* u_bf   = (unsigned short*)carve((size_t)LSEQ * HDIM * 2);
    float* BuRe = (float*)carve((size_t)LSEQ * PDIM * sizeof(float));
    float* BuIm = (float*)carve((size_t)LSEQ * PDIM * sizeof(float));
    float* car_re = (float*)carve((size_t)NBLK * PDIM * sizeof(float));
    float* car_im = (float*)carve((size_t)NBLK * PDIM * sizeof(float));
    float* st_re  = (float*)carve((size_t)NBLK * PDIM * sizeof(float));
    float* st_im  = (float*)carve((size_t)NBLK * PDIM * sizeof(float));
    unsigned short* xre_bf = (unsigned short*)carve((size_t)LSEQ * PDIM * 2);
    unsigned short* xim_bf = (unsigned short*)carve((size_t)LSEQ * PDIM * 2);

    // 1) discretization + operand precompute (all bandwidth-trivial)
    s5_disc_kernel<<<1, PDIM, 0, stream>>>(Lre, Lim, log_step, lam_re, lam_im, fac_re, fac_im);
    s5_u2bf_kernel<<<(LSEQ * HDIM / 8) / 256, 256, 0, stream>>>(u, u_bf);
    s5_bbar_kernel<<<(PDIM * HDIM + 255) / 256, 256, 0, stream>>>(Bre, Bim, fac_re, fac_im, Bre_bf, Bim_bf);
    s5_cbf_kernel<<<(HDIM * PDIM + 255) / 256, 256, 0, stream>>>(Cre, Cim, Cre_bf, Cng_bf);
    // 2) GEMM1: Bu (re, im)
    {
        dim3 grid(LSEQ / 512, PDIM / 16);
        s5_gemm1_kernel<<<grid, 256, 0, stream>>>(u_bf, Bre_bf, Bim_bf, BuRe, BuIm);
    }
    // 3) blocked scan
    s5_scan1_kernel<<<(PDIM * NBLK) / 256, 256, 0, stream>>>(lam_re, lam_im, BuRe, BuIm, car_re, car_im);
    s5_scan2_kernel<<<(PDIM + 255) / 256, 256, 0, stream>>>(lam_re, lam_im, car_re, car_im, st_re, st_im);
    s5_scan3_kernel<<<(PDIM * NBLK) / 256, 256, 0, stream>>>(lam_re, lam_im, BuRe, BuIm, st_re, st_im, xre_bf, xim_bf);
    // 4) GEMM2 + feedthrough
    {
        dim3 grid(LSEQ / 512, HDIM / 16);
        s5_gemm2_kernel<<<grid, 256, 0, stream>>>(xre_bf, xim_bf, Cre_bf, Cng_bf, u, Dvec, out);
    }
    (void)in_sizes; (void)n_in; (void)out_size; (void)ws_size;
}
